// RiemannFMDualStreamCross_20693152432566
// MI455X (gfx1250) — matrix operand order repaired and checked by reading, not compiled, not run on hardware
//
#include <hip/hip_runtime.h>
#include <hip/hip_bf16.h>

#define Bb 4
#define Nn 256
#define Dd 256
#define Ee 128
#define MN (Bb*Nn)

typedef __attribute__((ext_vector_type(16))) __bf16 v16bf;
typedef __attribute__((ext_vector_type(8)))  __bf16 v8bf;
typedef __attribute__((ext_vector_type(8)))  float  v8f;

// ---- fragment helpers (layouts per CDNA5 ISA 7.12.2, wave32) ----
// A (16x32 bf16, row-major source): lane r=lane&15 holds row r; half=lane>>4
// selects K half: elements 0-15 = K[half*16 .. +15] (contiguous, matches B).
__device__ __forceinline__ v16bf load_a_frag(const __bf16* row, int k0, int kh) {
  v8bf lo = *(const v8bf*)(row + k0 + kh * 16);
  v8bf hi = *(const v8bf*)(row + k0 + kh * 16 + 8);
  return __builtin_shufflevector(lo, hi, 0,1,2,3,4,5,6,7,8,9,10,11,12,13,14,15);
}
// B (32x16 bf16) from W^T stored (N x K) row-major: lane holds column n=lane&15,
// K half (lane>>4)*16, 16 contiguous bf16 = one 32B load.
__device__ __forceinline__ v16bf load_b_frag(const __bf16* Bt, int n, int K, int k0, int kh) {
  return *(const v16bf*)(Bt + (size_t)n * K + k0 + kh * 16);
}
__device__ __forceinline__ v8f wmma_bf16(v16bf a, v16bf b, v8f c) {
  return __builtin_amdgcn_wmma_f32_16x16x32_bf16(false, a, false, b, (short)0, c, false, false);
}
__device__ __forceinline__ float silu_f(float x) { return x / (1.0f + __expf(-x)); }

// ---------------- small prep kernels ----------------
__global__ void cvt_bf16_kernel(const float* __restrict__ x, __bf16* __restrict__ y, int n) {
  int i = blockIdx.x * blockDim.x + threadIdx.x;
  if (i < n) y[i] = (__bf16)x[i];
}

__global__ void transpose_bf16_kernel(const float* __restrict__ W, __bf16* __restrict__ Wt,
                                      int R, int C) {
  int i = blockIdx.x * blockDim.x + threadIdx.x;
  if (i < R * C) {
    int r = i / C, c = i % C;
    Wt[(size_t)c * R + r] = (__bf16)W[i];
  }
}

// Mt[e*D + d] = bf16( (1/sqrt(D)) * dot(Wq[d,:], Wk[e,:]) )   (folds Wq@Wk^T and scale)
__global__ void mt_kernel(const float* __restrict__ Wq, const float* __restrict__ Wk,
                          __bf16* __restrict__ Mt) {
  int e = blockIdx.x;
  int d = threadIdx.x;
  float s = 0.f;
  const float* wq = Wq + (size_t)d * Dd;
  const float* wk = Wk + (size_t)e * Dd;
  for (int x = 0; x < Dd; ++x) s += wq[x] * wk[x];
  Mt[(size_t)e * Dd + d] = (__bf16)(s * 0.0625f); // 1/sqrt(256)
}

// ---------------- generic wave-per-tile WMMA GEMM:  C[MxN] = A[MxK] * Bt[NxK]^T ----------------
constexpr int F_BIAS = 1, F_SILU = 2, F_RESID = 4, F_CF = 8, F_CB = 16;

template <int FLAGS>
__global__ __launch_bounds__(32) void gemm16_kernel(
    const __bf16* __restrict__ A, const __bf16* __restrict__ Bt, int N, int K,
    const float* __restrict__ bias, const float* __restrict__ resid,
    float* __restrict__ Cf, __bf16* __restrict__ Cb) {
  const int n0 = blockIdx.x * 16, m0 = blockIdx.y * 16;
  const int lane = threadIdx.x;
  const int r = lane & 15, kh = lane >> 4;
  v8f acc = {};
  const __bf16* arow = A + (size_t)(m0 + r) * K;
  for (int k0 = 0; k0 < K; k0 += 32) {
    v16bf a = load_a_frag(arow, k0, kh);
    v16bf b = load_b_frag(Bt, n0 + r, K, k0, kh);
    acc = wmma_bf16(a, b, acc);
  }
  const int n = n0 + r, moff = kh * 8;
  for (int v = 0; v < 8; ++v) {
    int m = m0 + moff + v;
    float x = acc[v];
    if (FLAGS & F_BIAS)  x += bias[n];
    if (FLAGS & F_SILU)  x = silu_f(x);
    if (FLAGS & F_RESID) x += resid[(size_t)m * N + n];
    if (FLAGS & F_CF)    Cf[(size_t)m * N + n] = x;
    if (FLAGS & F_CB)    Cb[(size_t)m * N + n] = (__bf16)x;
  }
}

// ---------------- attention: scores via g.t_i, softmax, u = sum_j alpha*g ----------------
__global__ __launch_bounds__(256) void attn_kernel(
    const float* __restrict__ g, const float* __restrict__ t,
    const unsigned char* __restrict__ mask, __bf16* __restrict__ u_bf) {
  const int i = blockIdx.x, b = blockIdx.y, tid = threadIdx.x;
  __shared__ __align__(16) float tloc[Ee];
  __shared__ float red[256];
  __shared__ float alpha[Nn];

  if (tid < Ee) tloc[tid] = t[((size_t)b * Nn + i) * Ee + tid];
  __syncthreads();

  const bool maski = mask[b * Nn + i] != 0;
  // pass 1: one j per thread, 512B contiguous row read per thread
  const int j = tid;
  const float4* grow = (const float4*)(g + (((size_t)b * Nn + i) * Nn + j) * Ee);
  const float4* tv4 = (const float4*)tloc;
  float s = 0.f;
  for (int x = 0; x < Ee / 4; ++x) {
    float4 gv = grow[x], tv = tv4[x];
    s += gv.x * tv.x + gv.y * tv.y + gv.z * tv.z + gv.w * tv.w;
  }
  if (!(maski && (mask[b * Nn + j] != 0))) s = -10000.0f;

  red[tid] = s; __syncthreads();
  for (int st = 128; st > 0; st >>= 1) { if (tid < st) red[tid] = fmaxf(red[tid], red[tid + st]); __syncthreads(); }
  const float mx = red[0]; __syncthreads();
  const float ev = __expf(s - mx);
  red[tid] = ev; __syncthreads();
  for (int st = 128; st > 0; st >>= 1) { if (tid < st) red[tid] += red[tid + st]; __syncthreads(); }
  const float sum = red[0]; __syncthreads();
  alpha[tid] = maski ? (ev / sum) : 0.0f;
  __syncthreads();

  // pass 2: u[e] = sum_j alpha[j]*g[...,j,e]; two coalesced j-halves
  const int e = tid & (Ee - 1);
  const int jbase = (tid >> 7) * 128;
  const float* gbase = g + (((size_t)b * Nn + i) * Nn + jbase) * Ee + e;
  float acc = 0.f;
  for (int jj = 0; jj < 128; ++jj) acc += alpha[jbase + jj] * gbase[(size_t)jj * Ee];
  red[tid] = acc; __syncthreads();
  if (tid < Ee) u_bf[((size_t)b * Nn + i) * Ee + tid] = (__bf16)(red[tid] + red[tid + 128]);
}

// ---------------- edge update: pre = xi+xj+had@Wh+b1b ; g_out = g + silu(pre)@W2b + b2b ----------------
__global__ __launch_bounds__(256) void edge_kernel(
    const float* __restrict__ g, const __bf16* __restrict__ hu_bf,
    const float* __restrict__ xi, const float* __restrict__ xj,
    const float* __restrict__ b1b, const float* __restrict__ b2b,
    const __bf16* __restrict__ WhT, const __bf16* __restrict__ W2bT,
    float* __restrict__ gout) {
  const int jt = blockIdx.x, i = blockIdx.y, b = blockIdx.z;
  const int j0 = jt * 32, tid = threadIdx.x;

  __shared__ __bf16 hu_i[Dd];                 // 512 B
  __shared__ __align__(16) __bf16 had[32][Dd]; // 16 KB  bf16(hu_i[d]*hu_j[d])
  __shared__ float xi_s[Ee], b1b_s[Ee], b2b_s[Ee];
  __shared__ float xj_s[32][Ee];              // 16 KB
  __shared__ __align__(16) __bf16 P[32][Ee];  // 8 KB   silu(pre)

  const size_t gtile = (((size_t)b * Nn + i) * Nn + j0) * Ee;
  // warm L2 for the g tile re-read in the epilogue (global_prefetch_b8)
  if (tid < 128) __builtin_prefetch(g + gtile + (size_t)tid * 32, 0, 1);

  if (tid < Dd) hu_i[tid] = hu_bf[((size_t)b * Nn + i) * Dd + tid];
  if (tid < Ee) {
    xi_s[tid]  = xi[((size_t)b * Nn + i) * Ee + tid];
    b1b_s[tid] = b1b[tid];
    b2b_s[tid] = b2b[tid];
  }
  __syncthreads();
  for (int idx = tid; idx < 32 * Dd; idx += 256) {
    int r = idx >> 8, d = idx & (Dd - 1);
    had[r][d] = (__bf16)((float)hu_bf[((size_t)b * Nn + j0 + r) * Dd + d] * (float)hu_i[d]);
  }
  for (int idx = tid; idx < 32 * Ee; idx += 256) {
    int r = idx >> 7, e = idx & (Ee - 1);
    xj_s[r][e] = xj[((size_t)b * Nn + j0 + r) * Ee + e];
  }
  __syncthreads();

  const int w = tid >> 5, lane = tid & 31;
  const int r = lane & 15, kh = lane >> 4, nb = w * 16;

  // GEMM1: pre_acc[32 x 16cols] = had(32x256) @ Wh(256x128), this wave's 16-col slice
  v8f c0 = {}, c1 = {};
  for (int kk = 0; kk < 8; ++kk) {
    const int k0 = kk * 32;
    v16bf a0 = load_a_frag(&had[r][0], k0, kh);
    v16bf a1 = load_a_frag(&had[r + 16][0], k0, kh);
    v16bf bf = load_b_frag(WhT, nb + r, Dd, k0, kh);
    c0 = wmma_bf16(a0, bf, c0);
    c1 = wmma_bf16(a1, bf, c1);
  }

  // epilogue: + xi + xj + b1b, silu, transpose C-layout -> A-layout via LDS
  const int e = nb + r, moff = kh * 8;
  for (int v = 0; v < 8; ++v) {
    int m0 = moff + v, m1 = 16 + moff + v;
    float p0 = c0[v] + xi_s[e] + xj_s[m0][e] + b1b_s[e];
    float p1 = c1[v] + xi_s[e] + xj_s[m1][e] + b1b_s[e];
    P[m0][e] = (__bf16)silu_f(p0);
    P[m1][e] = (__bf16)silu_f(p1);
  }
  __syncthreads();

  // GEMM2: out2[32 x 16cols] = P(32x128) @ W2b(128x128)
  v8f d0 = {}, d1 = {};
  for (int kk = 0; kk < 4; ++kk) {
    const int k0 = kk * 32;
    v16bf a0 = load_a_frag(&P[r][0], k0, kh);
    v16bf a1 = load_a_frag(&P[r + 16][0], k0, kh);
    v16bf bf = load_b_frag(W2bT, nb + r, Ee, k0, kh);
    d0 = wmma_bf16(a0, bf, d0);
    d1 = wmma_bf16(a1, bf, d1);
  }

  // g_update = g + out2 + b2b
  for (int v = 0; v < 8; ++v) {
    int m0 = moff + v, m1 = 16 + moff + v;
    size_t i0 = gtile + (size_t)m0 * Ee + e;
    size_t i1 = gtile + (size_t)m1 * Ee + e;
    gout[i0] = g[i0] + d0[v] + b2b_s[e];
    gout[i1] = g[i1] + d1[v] + b2b_s[e];
  }
}

// ---------------- launch ----------------
extern "C" void kernel_launch(void* const* d_in, const int* in_sizes, int n_in,
                              void* d_out, int out_size, void* d_ws, size_t ws_size,
                              hipStream_t stream) {
  const float* h    = (const float*)d_in[0];
  const float* g    = (const float*)d_in[1];
  const unsigned char* mask = (const unsigned char*)d_in[2];
  const float* Wq   = (const float*)d_in[3];
  const float* Wk   = (const float*)d_in[4];
  const float* Wv   = (const float*)d_in[5];
  const float* W1a  = (const float*)d_in[6];
  const float* b1a  = (const float*)d_in[7];
  const float* W2a  = (const float*)d_in[8];
  const float* b2a  = (const float*)d_in[9];
  const float* W1b  = (const float*)d_in[10];
  const float* b1b  = (const float*)d_in[11];
  const float* W2b  = (const float*)d_in[12];
  const float* b2b  = (const float*)d_in[13];

  float* out_h = (float*)d_out;                       // (B,N,D)
  float* out_g = (float*)d_out + (size_t)MN * Dd;     // (B,N,N,E)

  char* w = (char*)d_ws; size_t off = 0;
  auto alloc = [&](size_t bytes) { void* p = w + off; off = (off + bytes + 255) & ~(size_t)255; return p; };
  __bf16* h_bf  = (__bf16*)alloc((size_t)MN * Dd * 2);
  __bf16* Mt    = (__bf16*)alloc((size_t)Ee * Dd * 2);
  __bf16* WvT   = (__bf16*)alloc((size_t)Dd * Ee * 2);
  __bf16* W1aT  = (__bf16*)alloc((size_t)Dd * Dd * 2);
  __bf16* W2aT  = (__bf16*)alloc((size_t)Dd * Dd * 2);
  __bf16* WiT   = (__bf16*)alloc((size_t)Ee * Dd * 2);
  __bf16* WjT   = (__bf16*)alloc((size_t)Ee * Dd * 2);
  __bf16* WhT   = (__bf16*)alloc((size_t)Ee * Dd * 2);
  __bf16* W2bT  = (__bf16*)alloc((size_t)Ee * Ee * 2);
  float*  t     = (float*) alloc((size_t)MN * Ee * 4);
  __bf16* u_bf  = (__bf16*)alloc((size_t)MN * Ee * 2);
  __bf16* na_bf = (__bf16*)alloc((size_t)MN * Dd * 2);
  __bf16* z1_bf = (__bf16*)alloc((size_t)MN * Dd * 2);
  __bf16* hu_bf = (__bf16*)alloc((size_t)MN * Dd * 2);
  float*  xi    = (float*) alloc((size_t)MN * Ee * 4);
  float*  xj    = (float*) alloc((size_t)MN * Ee * 4);

  // weight prep
  cvt_bf16_kernel<<<(MN * Dd + 255) / 256, 256, 0, stream>>>(h, h_bf, MN * Dd);
  transpose_bf16_kernel<<<(Ee * Dd + 255) / 256, 256, 0, stream>>>(Wv, WvT, Ee, Dd);
  transpose_bf16_kernel<<<(Dd * Dd + 255) / 256, 256, 0, stream>>>(W1a, W1aT, Dd, Dd);
  transpose_bf16_kernel<<<(Dd * Dd + 255) / 256, 256, 0, stream>>>(W2a, W2aT, Dd, Dd);
  transpose_bf16_kernel<<<(Dd * Ee + 255) / 256, 256, 0, stream>>>(W1b,             WiT, Dd, Ee);
  transpose_bf16_kernel<<<(Dd * Ee + 255) / 256, 256, 0, stream>>>(W1b + Dd * Ee,   WjT, Dd, Ee);
  transpose_bf16_kernel<<<(Dd * Ee + 255) / 256, 256, 0, stream>>>(W1b + 2 * Dd * Ee, WhT, Dd, Ee);
  transpose_bf16_kernel<<<(Ee * Ee + 255) / 256, 256, 0, stream>>>(W2b, W2bT, Ee, Ee);
  mt_kernel<<<Ee, Dd, 0, stream>>>(Wq, Wk, Mt);

  // t = h @ (Wq Wk^T * scale)
  gemm16_kernel<F_CF><<<dim3(Ee / 16, MN / 16), 32, 0, stream>>>(h_bf, Mt, Ee, Dd, nullptr, nullptr, t, nullptr);
  // attention -> u (bf16)
  attn_kernel<<<dim3(Nn, Bb), 256, 0, stream>>>(g, t, mask, u_bf);
  // node_agg = u @ Wv (bf16 out)
  gemm16_kernel<F_CB><<<dim3(Dd / 16, MN / 16), 32, 0, stream>>>(u_bf, WvT, Dd, Ee, nullptr, nullptr, nullptr, na_bf);
  // z1 = silu(na @ W1a + b1a)
  gemm16_kernel<F_BIAS | F_SILU | F_CB><<<dim3(Dd / 16, MN / 16), 32, 0, stream>>>(na_bf, W1aT, Dd, Dd, b1a, nullptr, nullptr, z1_bf);
  // h_update = h + z1 @ W2a + b2a   (f32 -> d_out, bf16 -> hu_bf)
  gemm16_kernel<F_BIAS | F_RESID | F_CF | F_CB><<<dim3(Dd / 16, MN / 16), 32, 0, stream>>>(z1_bf, W2aT, Dd, Dd, b2a, h, out_h, hu_bf);
  // xi = hu @ Wi ; xj = hu @ Wj
  gemm16_kernel<F_CF><<<dim3(Ee / 16, MN / 16), 32, 0, stream>>>(hu_bf, WiT, Ee, Dd, nullptr, nullptr, xi, nullptr);
  gemm16_kernel<F_CF><<<dim3(Ee / 16, MN / 16), 32, 0, stream>>>(hu_bf, WjT, Ee, Dd, nullptr, nullptr, xj, nullptr);
  // edge update
  edge_kernel<<<dim3(Nn / 32, Nn, Bb), 256, 0, stream>>>(g, hu_bf, xi, xj, b1b, b2b, WhT, W2bT, out_g);

  (void)in_sizes; (void)n_in; (void)out_size; (void)ws_size;
}
